// KUCNet_trans_34840774705590
// MI455X (gfx1250) — matrix operand order repaired and checked by reading, not compiled, not run on hardware
//
#include <hip/hip_runtime.h>
#include <hip/hip_bf16.h>

// ===========================================================================
// KUCNet hot path on gfx1250 (MI455X, wave32, WMMA).
//
// Pipeline (all on `stream`, workspace carved from d_ws):
//   1. cvt f32->bf16: hidden, h0, Ws, W_h, W_ih, W_hh
//   2. tiny tables: rel_att[R,A]=rel_emb@Wr.T ; q_att[B,A]=rel_emb[q_rel]@Wqr.T+b_qr
//   3. WMMA GEMM: node_att[N,A] = hidden_bf @ Ws_bf.T           (f32 out)
//   4. edge kernel: alpha via table lookups + shfl reduce (+ global_prefetch
//      of the gathered rows); msg scatter with global_atomic_add_f32
//   5. cvt agg -> bf16 ; WMMA GEMM h_new = relu(agg @ W_h.T)    (bf16 out)
//   6. fused GRU+score: gi = h_new@W_ih.T+b_ih, gh = h0@W_hh.T+b_hh (WMMA into
//      LDS, A-fragments hoisted into registers), gates, h_out.W_final -> scores
// ===========================================================================

typedef __bf16 bf16_t;
typedef __attribute__((ext_vector_type(16))) __bf16 v16bf;
typedef __attribute__((ext_vector_type(8)))  __bf16 v8bf;
typedef __attribute__((ext_vector_type(8)))  float  v8f;

#define D_DIM 128   // hidden dim == GEMM K (structural)
#define A_DIM 64    // attention dim
#define G_DIM 384   // 3*D (GRU gates)
#define KFRAG (D_DIM / 32)   // 4 WMMA K-steps

// ---------------------------------------------------------------- helpers ---
__device__ __forceinline__ unsigned short f32_bf16(float f) {
    union { float f; unsigned u; } x; x.f = f;
    unsigned u = x.u;
    u += 0x7fffu + ((u >> 16) & 1u);   // round-to-nearest-even
    return (unsigned short)(u >> 16);
}

__device__ __forceinline__ float sigmoidf_(float x) {
    return 1.f / (1.f + __expf(-x));
}

// A-matrix fragment, 16x32 bf16, wave32 (ISA 7.12.2):
//   lanes 0-15: M=lane,   VGPR0-3 K=k0+{0..7},    VGPR4-7 K=k0+16+{0..7}
//   lanes16-31: M=lane-16, same with +8 K offset
__device__ __forceinline__ v16bf load_a_frag(const bf16_t* base, int ld,
                                             long row, int k0, int half) {
    const bf16_t* p = base + row * (long)ld + k0 + half * 8;
    union { v16bf v; v8bf h[2]; } u;
    u.h[0] = *(const v8bf*)(p);        // K = k0+half*8 + 0..7
    u.h[1] = *(const v8bf*)(p + 16);   // K = k0+16+half*8 + 0..7
    return u.v;
}

// B-matrix fragment, 32x16 bf16: lane n holds column N=(lane&15), K values
// k0+(lane>>4)*8+{0..7} and k0+16+(lane>>4)*8+{0..7}. W is row-major [Nc,K]
// (out = act @ W.T, so B column n == W row n).
__device__ __forceinline__ v16bf load_b_frag(const bf16_t* base, int ld,
                                             long row, int k0, int half) {
    const bf16_t* p = base + row * (long)ld + k0 + half * 8;
    union { v16bf v; v8bf h[2]; } u;
    u.h[0] = *(const v8bf*)(p);
    u.h[1] = *(const v8bf*)(p + 16);
    return u.v;
}

// ------------------------------------------------------------ tiny kernels --
__global__ void k_zero(float* p, long n) {
    long i = (long)blockIdx.x * blockDim.x + threadIdx.x;
    if (i < n) p[i] = 0.f;
}

__global__ void k_cvt_bf16(const float* s, unsigned short* d, long n) {
    long i = (long)blockIdx.x * blockDim.x + threadIdx.x;
    if (i < n) d[i] = f32_bf16(s[i]);
}

// rel_att[r,a] = rel_emb[r] . Wr[a] ; qrel_att[r,a] = rel_emb[r] . Wqr[a]
__global__ void k_rel_tables(const float* __restrict__ rel_emb,
                             const float* __restrict__ Wr,
                             const float* __restrict__ Wqr,
                             float* rel_att, float* qrel_att, int R) {
    int idx = blockIdx.x * blockDim.x + threadIdx.x;
    if (idx >= R * A_DIM) return;
    int r = idx / A_DIM, a = idx % A_DIM;
    const float* e  = rel_emb + (long)r * D_DIM;
    const float* wr = Wr  + (long)a * D_DIM;
    const float* wq = Wqr + (long)a * D_DIM;
    float s0 = 0.f, s1 = 0.f;
    #pragma unroll 4
    for (int k = 0; k < D_DIM; k++) { float ev = e[k]; s0 += ev * wr[k]; s1 += ev * wq[k]; }
    rel_att[idx]  = s0;
    qrel_att[idx] = s1;
}

// q_att[b,a] = qrel_att[q_rel[b],a] + b_qr[a]
__global__ void k_q_att(const float* __restrict__ qrel_att,
                        const int* __restrict__ q_rel,
                        const float* __restrict__ b_qr,
                        float* q_att, int B) {
    int idx = blockIdx.x * blockDim.x + threadIdx.x;
    if (idx >= B * A_DIM) return;
    int b = idx / A_DIM, a = idx % A_DIM;
    q_att[idx] = qrel_att[(long)q_rel[b] * A_DIM + a] + b_qr[a];
}

// --------------------------------------------------------- WMMA dense GEMM --
// C[M,Nc] = act[M,K=128](bf16) @ W[Nc,128](bf16)^T.
// One 16-row tile per block; each wave hoists the 4 A-fragments (16x128 tile)
// into registers once, then sweeps NT n-tiles against them (2 B-loads + 1 WMMA
// per inner step). Block = (Nc/(16*NT)) waves.
template <int RELU, int OUT_BF16, int NT>
__global__ void k_gemm_wmma(const unsigned short* __restrict__ A_,
                            const unsigned short* __restrict__ W_,
                            void* __restrict__ out_, long M, int Nc) {
    const bf16_t* A = (const bf16_t*)A_;
    const bf16_t* W = (const bf16_t*)W_;
    const int lane = threadIdx.x & 31;
    const int wave = threadIdx.x >> 5;
    const int lrow = lane & 15, half = lane >> 4;
    const long m0 = (long)blockIdx.x * 16;
    long arow = m0 + lrow; if (arow >= M) arow = M - 1;   // clamp, uniform-safe

    v16bf afr[KFRAG];
    #pragma unroll
    for (int k = 0; k < KFRAG; k++)
        afr[k] = load_a_frag(A, D_DIM, arow, k * 32, half);

    #pragma unroll
    for (int t = 0; t < NT; t++) {
        const int n0 = (wave * NT + t) * 16;
        const long brow = n0 + lrow;
        v8f acc = {};
        #pragma unroll
        for (int k = 0; k < KFRAG; k++) {
            v16bf bf = load_b_frag(W, D_DIM, brow, k * 32, half);
            acc = __builtin_amdgcn_wmma_f32_16x16x32_bf16(
                      false, afr[k], false, bf, (short)0, acc, false, false);
        }
        #pragma unroll
        for (int i = 0; i < 8; i++) {
            long row = m0 + i + 8 * half;   // C/D layout: VGPR i -> row i+8*half
            int  col = n0 + lrow;           //              col = lane&15
            if (row < M) {
                float v = acc[i];
                if (RELU) v = v > 0.f ? v : 0.f;
                if (OUT_BF16) ((unsigned short*)out_)[row * Nc + col] = f32_bf16(v);
                else          ((float*)out_)[row * Nc + col] = v;
            }
        }
    }
}

// ------------------------------------------------------------- edge kernel --
// One wave per edge. att = relu(node_att[sub] + rel_att[rel] + q_att[batch]),
// alpha = sigmoid(att.w_alpha + b_alpha), scatter alpha*(hs+hr) into agg[obj].
// hidden/rel_emb rows are prefetched (global_prefetch_b8) while the attention
// dot + wave reduction is in flight.
__global__ void __launch_bounds__(128) k_edges(
        const float* __restrict__ node_att, const float* __restrict__ rel_att,
        const float* __restrict__ q_att,    const float* __restrict__ w_alpha,
        const float* __restrict__ b_alpha,  const float* __restrict__ hidden,
        const float* __restrict__ rel_emb,
        const int* __restrict__ sub, const int* __restrict__ rel,
        const int* __restrict__ obj, const int* __restrict__ ebat,
        float* agg, long E) {
    const int wave = threadIdx.x >> 5;
    const int lane = threadIdx.x & 31;
    const long e = (long)blockIdx.x * 4 + wave;
    if (e >= E) return;
    const int s = sub[e], r = rel[e], o = obj[e], b = ebat[e];

    const float* hs = hidden  + (long)s * D_DIM;
    const float* hr = rel_emb + (long)r * D_DIM;
    // warm the gathered message rows (4 x 128B lines each) before they're used
    __builtin_prefetch(hs + lane * 4, 0, 3);
    __builtin_prefetch(hr + lane * 4, 0, 3);

    const float* na = node_att + (long)s * A_DIM;
    const float* ra = rel_att  + (long)r * A_DIM;
    const float* qa = q_att    + (long)b * A_DIM;
    float a0 = na[lane]      + ra[lane]      + qa[lane];
    float a1 = na[lane + 32] + ra[lane + 32] + qa[lane + 32];
    a0 = a0 > 0.f ? a0 : 0.f;
    a1 = a1 > 0.f ? a1 : 0.f;
    float part = a0 * w_alpha[lane] + a1 * w_alpha[lane + 32];
    #pragma unroll
    for (int off = 16; off > 0; off >>= 1) part += __shfl_xor(part, off, 32);
    const float alpha = sigmoidf_(part + b_alpha[0]);

    float* ag = agg + (long)o * D_DIM;
    #pragma unroll
    for (int i = 0; i < 4; i++) {
        const int d = lane + 32 * i;
        atomicAdd(&ag[d], alpha * (hs[d] + hr[d]));  // global_atomic_add_f32
    }
}

// ----------------------------------------------------- fused GRU + scoring --
// Block = 128 threads (4 waves), one 16-row tile. Each wave hoists the A
// fragments of h_new and h0 (4 each) into registers, then computes 6 column
// tiles of gi and gh via WMMA into LDS, followed by elementwise gates and the
// per-row dot with W_final.
__global__ void __launch_bounds__(128) k_gru_score(
        const unsigned short* __restrict__ hnew_bf,
        const unsigned short* __restrict__ h0_bf,
        const unsigned short* __restrict__ Wih_bf,
        const unsigned short* __restrict__ Whh_bf,
        const float* __restrict__ b_ih, const float* __restrict__ b_hh,
        const float* __restrict__ h0_f32, const float* __restrict__ W_final,
        float* scores, long M) {
    __shared__ float sGi[16 * G_DIM];
    __shared__ float sGh[16 * G_DIM];
    __shared__ float sHo[16 * D_DIM];

    const int lane = threadIdx.x & 31;
    const int wave = threadIdx.x >> 5;
    const int lrow = lane & 15, half = lane >> 4;
    const long m0 = (long)blockIdx.x * 16;
    const bf16_t* Hn = (const bf16_t*)hnew_bf;
    const bf16_t* H0 = (const bf16_t*)h0_bf;
    const bf16_t* Wi = (const bf16_t*)Wih_bf;
    const bf16_t* Wh = (const bf16_t*)Whh_bf;
    long arow = m0 + lrow; if (arow >= M) arow = M - 1;

    v16bf an[KFRAG], a0f[KFRAG];
    #pragma unroll
    for (int k = 0; k < KFRAG; k++) {
        an[k]  = load_a_frag(Hn, D_DIM, arow, k * 32, half);
        a0f[k] = load_a_frag(H0, D_DIM, arow, k * 32, half);
    }

    for (int t = 0; t < 6; t++) {
        const int ct = wave * 6 + t;       // 0..23 -> column tile of [*,384]
        const int n0 = ct * 16;
        const long brow = n0 + lrow;
        v8f gi = {}, gh = {};
        #pragma unroll
        for (int k = 0; k < KFRAG; k++) {
            v16bf bi = load_b_frag(Wi, D_DIM, brow, k * 32, half);
            gi = __builtin_amdgcn_wmma_f32_16x16x32_bf16(
                     false, an[k], false, bi, (short)0, gi, false, false);
            v16bf bh = load_b_frag(Wh, D_DIM, brow, k * 32, half);
            gh = __builtin_amdgcn_wmma_f32_16x16x32_bf16(
                     false, a0f[k], false, bh, (short)0, gh, false, false);
        }
        #pragma unroll
        for (int i = 0; i < 8; i++) {
            const int row = i + 8 * half;
            const int col = n0 + lrow;
            sGi[row * G_DIM + col] = gi[i] + b_ih[col];
            sGh[row * G_DIM + col] = gh[i] + b_hh[col];
        }
    }
    __syncthreads();

    {   // gates: thread tid handles column tid of all 16 rows
        const int col = threadIdx.x;           // 0..127
        const float wf = W_final[col];
        for (int j = 0; j < 16; j++) {
            const long row = m0 + j;
            const float ir = sGi[j * G_DIM + col],        hrr = sGh[j * G_DIM + col];
            const float iz = sGi[j * G_DIM + 128 + col],  hz  = sGh[j * G_DIM + 128 + col];
            const float in_ = sGi[j * G_DIM + 256 + col], hn  = sGh[j * G_DIM + 256 + col];
            const float rg = sigmoidf_(ir + hrr);
            const float zg = sigmoidf_(iz + hz);
            const float ng = tanhf(in_ + rg * hn);
            const float h0v = (row < M) ? h0_f32[row * D_DIM + col] : 0.f;
            const float ho = (1.f - zg) * ng + zg * h0v;
            sHo[j * D_DIM + col] = ho * wf;
        }
    }
    __syncthreads();

    if (threadIdx.x < 16) {
        const long row = m0 + threadIdx.x;
        if (row < M) {
            float s = 0.f;
            for (int c = 0; c < D_DIM; c++) s += sHo[threadIdx.x * D_DIM + c];
            scores[row] = s;
        }
    }
}

// ------------------------------------------------------------------- host ---
extern "C" void kernel_launch(void* const* d_in, const int* in_sizes, int n_in,
                              void* d_out, int out_size, void* d_ws, size_t ws_size,
                              hipStream_t stream) {
    const float* hidden  = (const float*)d_in[0];
    const float* rel_emb = (const float*)d_in[1];
    const float* Ws      = (const float*)d_in[2];
    const float* Wr      = (const float*)d_in[3];
    const float* Wqr     = (const float*)d_in[4];
    const float* b_qr    = (const float*)d_in[5];
    const float* w_alpha = (const float*)d_in[6];
    const float* b_alpha = (const float*)d_in[7];
    const float* W_h     = (const float*)d_in[8];
    const float* W_ih    = (const float*)d_in[9];
    const float* W_hh    = (const float*)d_in[10];
    const float* b_ih    = (const float*)d_in[11];
    const float* b_hh    = (const float*)d_in[12];
    const float* W_final = (const float*)d_in[13];
    const float* h0      = (const float*)d_in[14];
    const int* sub  = (const int*)d_in[15];
    const int* rel  = (const int*)d_in[16];
    const int* obj  = (const int*)d_in[17];
    const int* ebat = (const int*)d_in[18];
    const int* qrel = (const int*)d_in[19];

    const long N = in_sizes[0] / D_DIM;
    const long R = in_sizes[1] / D_DIM;
    const long E = in_sizes[15];
    const long B = in_sizes[19];

    char* wsb = (char*)d_ws;
    size_t off = 0;
    auto carve = [&](size_t bytes) -> char* {
        char* p = wsb + off;
        off += (bytes + 255) & ~(size_t)255;
        return p;
    };
    unsigned short* hidden_bf = (unsigned short*)carve((size_t)N * D_DIM * 2);
    unsigned short* h0_bf     = (unsigned short*)carve((size_t)N * D_DIM * 2);
    unsigned short* hnew_bf   = (unsigned short*)carve((size_t)N * D_DIM * 2);
    unsigned short* agg_bf    = (unsigned short*)carve((size_t)N * D_DIM * 2);
    float*          agg       = (float*)carve((size_t)N * D_DIM * 4);
    float*          node_att  = (float*)carve((size_t)N * A_DIM * 4);
    unsigned short* Ws_bf     = (unsigned short*)carve((size_t)A_DIM * D_DIM * 2);
    unsigned short* Wh_bf     = (unsigned short*)carve((size_t)D_DIM * D_DIM * 2);
    unsigned short* Wih_bf    = (unsigned short*)carve((size_t)G_DIM * D_DIM * 2);
    unsigned short* Whh_bf    = (unsigned short*)carve((size_t)G_DIM * D_DIM * 2);
    float*          rel_att   = (float*)carve((size_t)R * A_DIM * 4);
    float*          qrel_att  = (float*)carve((size_t)R * A_DIM * 4);
    float*          q_att     = (float*)carve((size_t)B * A_DIM * 4);

    const long nND = N * D_DIM;
    const int  TB  = 256;
    auto blks = [](long n, int t) { return (unsigned)((n + t - 1) / t); };

    // 1. conversions + zero
    k_cvt_bf16<<<blks(nND, TB), TB, 0, stream>>>(hidden, hidden_bf, nND);
    k_cvt_bf16<<<blks(nND, TB), TB, 0, stream>>>(h0, h0_bf, nND);
    k_cvt_bf16<<<blks((long)A_DIM * D_DIM, TB), TB, 0, stream>>>(Ws, Ws_bf, (long)A_DIM * D_DIM);
    k_cvt_bf16<<<blks((long)D_DIM * D_DIM, TB), TB, 0, stream>>>(W_h, Wh_bf, (long)D_DIM * D_DIM);
    k_cvt_bf16<<<blks((long)G_DIM * D_DIM, TB), TB, 0, stream>>>(W_ih, Wih_bf, (long)G_DIM * D_DIM);
    k_cvt_bf16<<<blks((long)G_DIM * D_DIM, TB), TB, 0, stream>>>(W_hh, Whh_bf, (long)G_DIM * D_DIM);
    k_zero<<<blks(nND, TB), TB, 0, stream>>>(agg, nND);

    // 2. tiny relation / query tables
    k_rel_tables<<<blks(R * A_DIM, 128), 128, 0, stream>>>(rel_emb, Wr, Wqr,
                                                           rel_att, qrel_att, (int)R);
    k_q_att<<<blks(B * A_DIM, 128), 128, 0, stream>>>(qrel_att, qrel, b_qr, q_att, (int)B);

    // 3. node_att = hidden @ Ws.T   [N,64]: 1 wave/block, 4 n-tiles per wave
    k_gemm_wmma<0, 0, 4><<<blks(N, 16), 32, 0, stream>>>(hidden_bf, Ws_bf,
                                                         node_att, N, A_DIM);

    // 4. per-edge attention + message scatter (segment_sum via f32 atomics)
    k_edges<<<blks(E, 4), 128, 0, stream>>>(node_att, rel_att, q_att, w_alpha,
                                            b_alpha, hidden, rel_emb,
                                            sub, rel, obj, ebat, agg, E);

    // 5. h_new = relu(agg @ W_h.T)  [N,128]: 2 waves/block, 4 n-tiles per wave
    k_cvt_bf16<<<blks(nND, TB), TB, 0, stream>>>(agg, agg_bf, nND);
    k_gemm_wmma<1, 1, 4><<<blks(N, 16), 64, 0, stream>>>(agg_bf, Wh_bf,
                                                         hnew_bf, N, D_DIM);

    // 6. fused GRU gate + final scoring
    k_gru_score<<<blks(N, 16), 128, 0, stream>>>(hnew_bf, h0_bf, Wih_bf, Whh_bf,
                                                 b_ih, b_hh, h0, W_final,
                                                 (float*)d_out, N);
}
